// denoising_model_82377472737852
// MI455X (gfx1250) — compile-verified, matrix operand order checked by reading, not run on hardware
//
#include <hip/hip_runtime.h>
#include <hip/hip_bf16.h>
#include <math.h>

// ---------------------------------------------------------------------------
// Model dims
// ---------------------------------------------------------------------------
#define NN      50000
#define EE      800000
#define NFEAT   128
#define NHID    32
#define NHEAD   4
#define KP0     160    // pad of K=129 to multiple of 32
#define KP1     64     // pad of K=33 to multiple of 32
#define FOUT    66     // final hidden
#define FOUTP   80     // pad of 66 to multiple of 16
#define MTILES  (NN / 16)   // 3125, exact

typedef __attribute__((ext_vector_type(16))) _Float16 v16h;
typedef __attribute__((ext_vector_type(8)))  float    v8f;

// ---------------------------------------------------------------------------
// Helpers
// ---------------------------------------------------------------------------
__device__ __forceinline__ float eluf(float x) { return x > 0.f ? x : expm1f(x); }

__device__ __forceinline__ void atomicMaxF(float* addr, float val) {
    // monotonic ordered-int trick; works with -inf init
    if (val >= 0.f) atomicMax((int*)addr, __float_as_int(val));
    else            atomicMin((unsigned int*)addr, (unsigned int)__float_as_int(val));
}

__global__ void k_fill(float* p, float v, long long n) {
    long long i = (long long)blockIdx.x * blockDim.x + threadIdx.x;
    if (i < n) p[i] = v;
}

// ---------------------------------------------------------------------------
// WMMA GEMM:  C[M x Nc] = A_f16[M x Kpad] * Bt_f16[Nc x Kpad]^T + colbias[Nc]
// One wave computes a 16 x (16*NT) strip: A fragment loaded once per K-step,
// NT back-to-back WMMAs reuse it (8x less A traffic for the 128-col GEMM).
// M % 16 == 0, Nc == 16*NT, Kpad % 32 == 0.
// ---------------------------------------------------------------------------
#define GEMM_WPB 4
template<int NT>
__global__ __launch_bounds__(32 * GEMM_WPB)
void k_wmma_gemm(const _Float16* __restrict__ A,
                 const _Float16* __restrict__ Bt,
                 const float*    __restrict__ colbias,
                 float*          __restrict__ C,
                 int Mtiles, int Kpad) {
    const int lane  = threadIdx.x & 31;
    const int wave  = threadIdx.x >> 5;
    const int mtile = blockIdx.x * GEMM_WPB + wave;
    if (mtile >= Mtiles) return;                   // wave-uniform exit
    const int Nc = 16 * NT;

    const int rm     = lane & 15;
    const int kshift = (lane >> 4) << 3;           // 0 or 8

    const _Float16* ap = A + (size_t)(mtile * 16 + rm) * Kpad + kshift;

    v8f acc[NT] = {};
    for (int kk = 0; kk < Kpad; kk += 32) {
        union { v16h v; uint4 q[2]; } fa;
        fa.q[0] = *(const uint4*)(ap);             // k: base+0..7
        fa.q[1] = *(const uint4*)(ap + 16);        // k: base+16..23
        __builtin_prefetch(ap + 32, 0, 1);         // global_prefetch_b8
#pragma unroll
        for (int tt = 0; tt < NT; ++tt) {
            const _Float16* bp = Bt + (size_t)(tt * 16 + rm) * Kpad + kk + kshift;
            union { v16h v; uint4 q[2]; } fb;
            fb.q[0] = *(const uint4*)(bp);
            fb.q[1] = *(const uint4*)(bp + 16);
            acc[tt] = __builtin_amdgcn_wmma_f32_16x16x32_f16(
                          false, fa.v, false, fb.v, (short)0, acc[tt], false, false);
        }
        ap += 32;
    }

    const int mrow = mtile * 16 + ((lane >> 4) << 3);
#pragma unroll
    for (int tt = 0; tt < NT; ++tt) {
        const int ccol = tt * 16 + (lane & 15);
        const float cb = colbias[ccol];
#pragma unroll
        for (int v = 0; v < 8; ++v)
            C[(size_t)(mrow + v) * Nc + ccol] = acc[tt][v] + cb;
    }
}

// ---------------------------------------------------------------------------
// Weight prep: Bt[Rows x Kpad] f16 = transpose(W[K x Nw]) zero-padded
// ---------------------------------------------------------------------------
__global__ void k_prep_wt(const float* __restrict__ W, _Float16* __restrict__ Bt,
                          int K, int Nw, int Kpad, int Rows) {
    long long idx = (long long)blockIdx.x * blockDim.x + threadIdx.x;
    long long tot = (long long)Rows * Kpad;
    if (idx >= tot) return;
    int n = (int)(idx / Kpad), k = (int)(idx % Kpad);
    float v = (k < K && n < Nw) ? W[(size_t)k * Nw + n] : 0.f;
    Bt[idx] = (_Float16)v;
}

// h0 = concat(x, qY) -> f16 padded to KP0
__global__ void k_build_h0(const float* __restrict__ x, const float* __restrict__ qY,
                           _Float16* __restrict__ hA) {
    long long idx = (long long)blockIdx.x * blockDim.x + threadIdx.x;
    long long tot = (long long)NN * KP0;
    if (idx >= tot) return;
    int n = (int)(idx / KP0), k = (int)(idx % KP0);
    float v = (k < NFEAT) ? x[(size_t)n * NFEAT + k] : (k == NFEAT ? qY[n] : 0.f);
    hA[idx] = (_Float16)v;
}

// ---------------------------------------------------------------------------
// Time / positional embeddings + folded per-layer bias vectors (1 block)
// ---------------------------------------------------------------------------
__global__ void k_embed(const float* t_ptr,
                        const float* W1, const float* b1,
                        const float* W2, const float* b2,
                        const float* rW,
                        const float* gb0, const float* sb0,
                        const float* gb1, const float* sb1,
                        const float* gb2, const float* sb2,
                        const float* fb1,
                        float* t_rel, float* cvec, float* b1pad) {
    __shared__ float s_te[32], s_t1[64], s_pe[130], s_tabs[32];
    const float t = t_ptr[0];
    const int j = threadIdx.x;
    if (j < 16) {
        float f = __expf(-logf(10000.f) * (float)j / 15.f);
        float e = t * 4.f * f;
        s_te[j] = __sinf(e); s_te[j + 16] = __cosf(e);
    }
    __syncthreads();
    if (j < 64) {
        float acc = b1[j];
        for (int k = 0; k < 32; ++k) acc += s_te[k] * W1[k * 64 + j];
        s_t1[j] = eluf(acc);
    }
    __syncthreads();
    if (j < 32) {
        float acc = b2[j];
        for (int k = 0; k < 64; ++k) acc += s_t1[k] * W2[k * 32 + j];
        s_tabs[j] = acc;
    }
    if (j < 65) {
        float inv = __powf(10000.f, -(2.f * (float)j) / 129.f);
        float s = t * inv;
        s_pe[j] = __sinf(s); s_pe[65 + j] = __cosf(s);
    }
    __syncthreads();
    if (j < 128) {
        float acc = 0.f;
        for (int k = 0; k < 130; ++k) acc += s_pe[k] * rW[k * 128 + j];
        t_rel[j] = acc;
    }
    if (j < 32) {
        cvec[j]      = s_tabs[j] + gb0[j] + sb0[j];
        cvec[32 + j] = s_tabs[j] + gb1[j] + sb1[j];
        cvec[64 + j] = s_tabs[j] + gb2[j] + sb2[j];
    }
    if (j < FOUTP) b1pad[j] = (j < FOUT) ? fb1[j] : 0.f;
}

// a_src/a_dst: per (node, head) dot with attention vectors
__global__ void k_srcdst(const float* __restrict__ h4,
                         const float* __restrict__ att_src,
                         const float* __restrict__ att_dst,
                         float* __restrict__ a_src, float* __restrict__ a_dst) {
    int idx = blockIdx.x * blockDim.x + threadIdx.x;
    if (idx >= NN * NHEAD) return;
    int n = idx >> 2, h = idx & 3;
    const float* hp = h4 + (size_t)n * 128 + h * 32;
    const float* as = att_src + h * 32;
    const float* ad = att_dst + h * 32;
    float ss = 0.f, sd = 0.f;
    for (int c = 0; c < 32; ++c) { float v = hp[c]; ss += v * as[c]; sd += v * ad[c]; }
    a_src[idx] = ss; a_dst[idx] = sd;
}

// pass 1: edge logits + segment max (atomic)
__global__ void k_edge1(const int* __restrict__ src, const int* __restrict__ dst,
                        const float* __restrict__ a_src, const float* __restrict__ a_dst,
                        float* __restrict__ e4, float* __restrict__ emax) {
    long long idx = (long long)blockIdx.x * blockDim.x + threadIdx.x;
    if (idx >= (long long)EE * NHEAD) return;
    int e = (int)(idx >> 2), h = (int)(idx & 3);
    int s = src[e], d = dst[e];
    float v = a_src[s * 4 + h] + a_dst[d * 4 + h];
    v = v > 0.f ? v : 0.2f * v;                    // leaky_relu 0.2
    e4[idx] = v;
    atomicMaxF(&emax[d * 4 + h], v);
}

// pass 2: exp(e - max) + segment sum (atomic)
__global__ void k_edge2(const int* __restrict__ dst,
                        float* __restrict__ e4, const float* __restrict__ emax,
                        float* __restrict__ denom) {
    long long idx = (long long)blockIdx.x * blockDim.x + threadIdx.x;
    if (idx >= (long long)EE * NHEAD) return;
    int e = (int)(idx >> 2), h = (int)(idx & 3);
    int d = dst[e];
    float ex = __expf(e4[idx] - emax[d * 4 + h]);
    e4[idx] = ex;
    atomicAdd(&denom[d * 4 + h], ex);
}

// pass 3: alpha-weighted scatter-add of h4[src]
__global__ void k_edge3(const int* __restrict__ src, const int* __restrict__ dst,
                        const float* __restrict__ e4, const float* __restrict__ denom,
                        const float* __restrict__ h4, float* __restrict__ attn) {
    long long idx = (long long)blockIdx.x * blockDim.x + threadIdx.x;
    if (idx >= (long long)EE * 128) return;
    int e = (int)(idx >> 7), j = (int)(idx & 127), h = j >> 5;
    int s = src[e], d = dst[e];
    float alpha = e4[(size_t)e * 4 + h] / (denom[d * 4 + h] + 1e-16f);
    atomicAdd(&attn[(size_t)d * 128 + j], h4[(size_t)s * 128 + j] * alpha);
}

// combine: head-mean + residual bias-term + ELU; emit next padded f16 activations
__global__ void k_combine(const float* __restrict__ attn, const float* __restrict__ xb0,
                          const float* __restrict__ qY, _Float16* __restrict__ hA) {
    long long idx = (long long)blockIdx.x * blockDim.x + threadIdx.x;
    if (idx >= (long long)NN * KP1) return;
    int n = (int)(idx >> 6), c = (int)(idx & 63);
    _Float16 val;
    if (c < 32) {
        const float* ap = attn + (size_t)n * 128 + c;
        float m = 0.25f * (ap[0] + ap[32] + ap[64] + ap[96]);
        val = (_Float16)eluf(m + xb0[(size_t)n * 32 + c]);
    } else if (c == 32) {
        val = (_Float16)qY[n];
    } else {
        val = (_Float16)0.f;
    }
    hA[idx] = val;
}

// final: out[n,0:2] = elu(C1[n,:66]) @ W2 + b2
__global__ void k_final(const float* __restrict__ C1, const float* __restrict__ W2,
                        const float* __restrict__ b2, float* __restrict__ out) {
    int n = blockIdx.x * blockDim.x + threadIdx.x;
    if (n >= NN) return;
    float a0 = b2[0], a1 = b2[1];
    const float* cp = C1 + (size_t)n * FOUTP;
    for (int k = 0; k < FOUT; ++k) {
        float hv = eluf(cp[k]);
        a0 += hv * W2[k * 2];
        a1 += hv * W2[k * 2 + 1];
    }
    out[n * 2]     = a0;
    out[n * 2 + 1] = a1;
}

// ---------------------------------------------------------------------------
// Launch
// ---------------------------------------------------------------------------
extern "C" void kernel_launch(void* const* d_in, const int* in_sizes, int n_in,
                              void* d_out, int out_size, void* d_ws, size_t ws_size,
                              hipStream_t stream) {
    (void)in_sizes; (void)n_in; (void)out_size; (void)ws_size;

    const float* x   = (const float*)d_in[0];
    const float* qY  = (const float*)d_in[1];
    const int*   adj = (const int*)d_in[2];
    const float* t   = (const float*)d_in[3];
    // d_in[4] = num_steps (unused: t*4 folded)
    const float* fW1 = (const float*)d_in[5];
    const float* fW2 = (const float*)d_in[6];
    const float* fb1 = (const float*)d_in[7];
    const float* fb2 = (const float*)d_in[8];
    const float* gW[3]  = { (const float*)d_in[9],  (const float*)d_in[13], (const float*)d_in[17] };
    const float* gAd[3] = { (const float*)d_in[10], (const float*)d_in[14], (const float*)d_in[18] };
    const float* gAs[3] = { (const float*)d_in[11], (const float*)d_in[15], (const float*)d_in[19] };
    const float* gB[3]  = { (const float*)d_in[12], (const float*)d_in[16], (const float*)d_in[20] };
    const float* rW  = (const float*)d_in[21];
    const float* sW[3] = { (const float*)d_in[22], (const float*)d_in[24], (const float*)d_in[26] };
    const float* sB[3] = { (const float*)d_in[23], (const float*)d_in[25], (const float*)d_in[27] };
    const float* tW1 = (const float*)d_in[28];
    const float* tW2 = (const float*)d_in[29];
    const float* tb1 = (const float*)d_in[30];
    const float* tb2 = (const float*)d_in[31];

    const int* src = adj;
    const int* dst = adj + EE;

    // ---- workspace carve ----
    char* base = (char*)d_ws;
    size_t off = 0;
    auto alloc = [&](size_t bytes) -> void* {
        void* p = base + off;
        off = (off + bytes + 255) & ~(size_t)255;
        return p;
    };
    _Float16* hA    = (_Float16*)alloc((size_t)NN * KP0 * 2);
    float*    h4    = (float*)alloc((size_t)NN * 128 * 4);
    float*    attn  = (float*)alloc((size_t)NN * 128 * 4);
    float*    xb0   = (float*)alloc((size_t)NN * 32 * 4);
    float*    a_src = (float*)alloc((size_t)NN * 4 * 4);
    float*    a_dst = (float*)alloc((size_t)NN * 4 * 4);
    float*    emax  = (float*)alloc((size_t)NN * 4 * 4);
    float*    denom = (float*)alloc((size_t)NN * 4 * 4);
    float*    e4    = (float*)alloc((size_t)EE * 4 * 4);
    float*    C1    = (float*)alloc((size_t)NN * FOUTP * 4);
    _Float16* BtG0  = (_Float16*)alloc(128 * KP0 * 2);
    _Float16* BtG1  = (_Float16*)alloc(128 * KP1 * 2);
    _Float16* BtG2  = (_Float16*)alloc(128 * KP1 * 2);
    _Float16* BtS0  = (_Float16*)alloc(32 * KP0 * 2);
    _Float16* BtS1  = (_Float16*)alloc(32 * KP1 * 2);
    _Float16* BtS2  = (_Float16*)alloc(32 * KP1 * 2);
    _Float16* BtF1  = (_Float16*)alloc(FOUTP * KP1 * 2);
    float*    t_rel = (float*)alloc(128 * 4);
    float*    cvec  = (float*)alloc(96 * 4);
    float*    b1pad = (float*)alloc(FOUTP * 4);

    _Float16* BtG[3] = { BtG0, BtG1, BtG2 };
    _Float16* BtS[3] = { BtS0, BtS1, BtS2 };

    const int TB = 256;
    auto blocks = [&](long long n) { return (unsigned)((n + TB - 1) / TB); };

    // ---- embeddings + folded biases ----
    k_embed<<<1, 128, 0, stream>>>(t, tW1, tb1, tW2, tb2, rW,
                                   gB[0], sB[0], gB[1], sB[1], gB[2], sB[2], fb1,
                                   t_rel, cvec, b1pad);

    // ---- weight prep (transpose + pad + f16) ----
    k_prep_wt<<<blocks(128LL * KP0), TB, 0, stream>>>(gW[0], BtG0, 129, 128, KP0, 128);
    k_prep_wt<<<blocks(128LL * KP1), TB, 0, stream>>>(gW[1], BtG1, 33, 128, KP1, 128);
    k_prep_wt<<<blocks(128LL * KP1), TB, 0, stream>>>(gW[2], BtG2, 33, 128, KP1, 128);
    k_prep_wt<<<blocks(32LL * KP0),  TB, 0, stream>>>(sW[0], BtS0, 129, 32, KP0, 32);
    k_prep_wt<<<blocks(32LL * KP1),  TB, 0, stream>>>(sW[1], BtS1, 33, 32, KP1, 32);
    k_prep_wt<<<blocks(32LL * KP1),  TB, 0, stream>>>(sW[2], BtS2, 33, 32, KP1, 32);
    k_prep_wt<<<blocks((long long)FOUTP * KP1), TB, 0, stream>>>(fW1, BtF1, 33, FOUT, KP1, FOUTP);

    // ---- h0 = concat(x, qY) ----
    k_build_h0<<<blocks((long long)NN * KP0), TB, 0, stream>>>(x, qY, hA);

    const dim3 gemmBlk(32 * GEMM_WPB);
    const unsigned gemmGx = (MTILES + GEMM_WPB - 1) / GEMM_WPB;

    for (int i = 0; i < 3; ++i) {
        const int Kpad = (i == 0) ? KP0 : KP1;

        // h @ Wgat + t_rel  -> h4 [N,128]   (8 N-tiles per wave, A loaded once)
        k_wmma_gemm<8><<<gemmGx, gemmBlk, 0, stream>>>(hA, BtG[i], t_rel, h4,
                                                       MTILES, Kpad);
        // h @ Wskip + (t_abs + gat_bias + skip_b) -> xb0 [N,32]
        k_wmma_gemm<2><<<gemmGx, gemmBlk, 0, stream>>>(hA, BtS[i], cvec + i * 32, xb0,
                                                       MTILES, Kpad);
        // attention coefficients
        k_srcdst<<<blocks((long long)NN * 4), TB, 0, stream>>>(h4, gAs[i], gAd[i], a_src, a_dst);

        k_fill<<<blocks((long long)NN * 4), TB, 0, stream>>>(emax, -INFINITY, (long long)NN * 4);
        k_fill<<<blocks((long long)NN * 4), TB, 0, stream>>>(denom, 0.f, (long long)NN * 4);
        k_fill<<<blocks((long long)NN * 128), TB, 0, stream>>>(attn, 0.f, (long long)NN * 128);

        k_edge1<<<blocks((long long)EE * 4), TB, 0, stream>>>(src, dst, a_src, a_dst, e4, emax);
        k_edge2<<<blocks((long long)EE * 4), TB, 0, stream>>>(dst, e4, emax, denom);
        k_edge3<<<blocks((long long)EE * 128), TB, 0, stream>>>(src, dst, e4, denom, h4, attn);

        // head mean + residual + elu -> next hA [N, KP1] (col 32 = qY, rest 0)
        k_combine<<<blocks((long long)NN * KP1), TB, 0, stream>>>(attn, xb0, qY, hA);
    }

    // final MLP layer 1: hA [N,64] @ BtF1 -> C1 [N,80] (+b1)
    k_wmma_gemm<5><<<gemmGx, gemmBlk, 0, stream>>>(hA, BtF1, b1pad, C1, MTILES, KP1);
    // final MLP layer 2 (N=2 -> scalar)
    k_final<<<blocks(NN), TB, 0, stream>>>(C1, fW2, fb2, (float*)d_out);
}
